// MaskedMHSA_33612414058616
// MI455X (gfx1250) — matrix operand order repaired and checked by reading, not compiled, not run on hardware
//
#include <hip/hip_runtime.h>

#define BB 2
#define TT 2048
#define EE 768
#define HH 12
#define DHH 64
#define AW 4   // waves per attention block

typedef __attribute__((ext_vector_type(16))) __bf16 v16bf;
typedef __attribute__((ext_vector_type(8)))  float  v8f;
typedef __attribute__((ext_vector_type(4)))  unsigned int v4u;
typedef __attribute__((ext_vector_type(8)))  int v8i_t;
typedef __attribute__((ext_vector_type(4)))  int v4i_t;

// ---------------------------------------------------------------------------
// WMMA fragment helpers (CDNA5 wave32 layouts, CDNA5 ISA §7.12.2)
// ---------------------------------------------------------------------------

// A-matrix 16x32 bf16: lane L holds row M=L&15.
// vector elems 0..7  -> K = base + 0..7   (base = (L>=16) ? 8 : 0)
// vector elems 8..15 -> K = base + 16..23
static __device__ __forceinline__ v16bf load_A(const __bf16* base, int ld, int lane) {
    const __bf16* p = base + (size_t)(lane & 15) * ld + ((lane >> 4) << 3);
    v16bf a;
#pragma unroll
    for (int i = 0; i < 8; ++i) a[i] = p[i];
#pragma unroll
    for (int i = 0; i < 8; ++i) a[8 + i] = p[16 + i];
    return a;
}

// B-matrix 32x16 bf16: lane L holds col N=L&15; elems 0..7 -> K = base+0..7, elems 8..15 -> K = base+16..23 (base = (L>=16)?8:0).
// Caller provides a layout where (k,n) lives at n*ld + k (contiguous in k).
static __device__ __forceinline__ v16bf load_B(const __bf16* base, int ld, int lane) {
    const __bf16* p = base + (size_t)(lane & 15) * ld + ((lane >> 4) << 3);
    v16bf b;
#pragma unroll
    for (int i = 0; i < 8; ++i) b[i] = p[i];
#pragma unroll
    for (int i = 0; i < 8; ++i) b[8 + i] = p[16 + i];
    return b;
}

static __device__ __forceinline__ v8f wmma_bf16(v16bf a, v16bf b, v8f c) {
    return __builtin_amdgcn_wmma_f32_16x16x32_bf16(false, a, false, b, (short)0, c,
                                                   false, false);
}

// ---------------------------------------------------------------------------
// TDM: build D# descriptor groups per CDNA5 ISA and
// issue a 2D tile load Global -> LDS.  All inputs are wave-uniform.
//   tile_x   : elements per row of the tile  (dim0)
//   tile_y   : rows of the tile              (dim1)
//   stride0  : tensor dim0 stride (elements)
// Element size fixed at 2 bytes (bf16).
// Toolchain uses the 6-arg builtin: (g0 u32x4, g1 i32x8, g2 i32x4, g3 i32x4,
// extra i32x8, cpol i32).
// ---------------------------------------------------------------------------
static __device__ __forceinline__ void tdm_load_2d(
    unsigned lds_addr, unsigned long long gaddr,
    unsigned tile_x, unsigned tile_y,
    unsigned tensor_x, unsigned tensor_y, unsigned long long stride0) {
    v4u g0;
    g0[0] = 1u;                                               // count=1, user D#
    g0[1] = (unsigned)__builtin_amdgcn_readfirstlane((int)lds_addr);
    g0[2] = (unsigned)__builtin_amdgcn_readfirstlane((int)(gaddr & 0xffffffffull));
    g0[3] = ((unsigned)__builtin_amdgcn_readfirstlane(
                 (int)((gaddr >> 32) & 0x1ffffffull))) |
            (2u << 30);                                       // type=2 ("image")
    v8i_t g1;
    g1[0] = (int)(1u << 16);                                  // data_size=1 -> 2B
    g1[1] = (int)((tensor_x & 0xffffu) << 16);                // tensor_dim0 lo
    g1[2] = (int)((tensor_x >> 16) | ((tensor_y & 0xffffu) << 16));
    g1[3] = (int)((tensor_y >> 16) | (tile_x << 16));         // tile_dim0
    g1[4] = (int)(tile_y & 0xffffu);                          // tile_dim1 (dim2=0)
    g1[5] = (int)(unsigned)(stride0 & 0xffffffffull);         // dim0 stride lo
    g1[6] = (int)(unsigned)((stride0 >> 32) & 0xffffull);     // dim0 stride hi
    g1[7] = 0;
    v4i_t gz4 = {0, 0, 0, 0};
    v8i_t gz8 = {0, 0, 0, 0, 0, 0, 0, 0};
    __builtin_amdgcn_tensor_load_to_lds(g0, g1, gz4, gz4, gz8, 0);
}

// ---------------------------------------------------------------------------
// Stage 0: dtype conversion / weight pre-transposition
// ---------------------------------------------------------------------------

__global__ void cvt_x_kernel(const float* __restrict__ x, __bf16* __restrict__ xb, int n) {
    int i = blockIdx.x * blockDim.x + threadIdx.x;
    if (i < n) xb[i] = (__bf16)x[i];
}

// W [H,E,DH] -> Wt [H,DH,E]  (B-fragment friendly: contiguous in K=E)
__global__ void cvt_w_kernel(const float* __restrict__ W, __bf16* __restrict__ Wt) {
    int i = blockIdx.x * blockDim.x + threadIdx.x;
    if (i >= HH * EE * DHH) return;
    int h = i / (EE * DHH);
    int r = i % (EE * DHH);
    int e = r / DHH;
    int d = r % DHH;
    Wt[((size_t)h * DHH + d) * EE + e] = (__bf16)W[i];
}

// Wo [E,E] row-major [e_in][e_out] -> Wot [e_out][e_in]
__global__ void cvt_wo_kernel(const float* __restrict__ Wo, __bf16* __restrict__ Wot) {
    int i = blockIdx.x * blockDim.x + threadIdx.x;
    if (i >= EE * EE) return;
    int ein  = i / EE;
    int eout = i % EE;
    Wot[(size_t)eout * EE + ein] = (__bf16)Wo[i];
}

// ---------------------------------------------------------------------------
// Stage 1: per-head Q/K/V projections.  One wave per 16x16 output tile.
// grid = (T/16, B*H, 3), block = 128 (4 waves, one per 16-col ntile of DH=64)
// Q,K stored [B,H,T,DH]; V stored transposed [B,H,DH,T] for the PV B-frags.
// ---------------------------------------------------------------------------

__global__ __launch_bounds__(128) void qkv_proj_kernel(
    const __bf16* __restrict__ xb,
    const __bf16* __restrict__ Wtq, const __bf16* __restrict__ Wtk,
    const __bf16* __restrict__ Wtv,
    const float* __restrict__ bq, const float* __restrict__ bk,
    const float* __restrict__ bv,
    __bf16* __restrict__ Q, __bf16* __restrict__ K, __bf16* __restrict__ Vt) {
    const int lane = threadIdx.x & 31;
    const int wv   = threadIdx.x >> 5;   // ntile 0..3
    const int mt   = blockIdx.x;         // T/16 tiles
    const int bh   = blockIdx.y;
    const int z    = blockIdx.z;         // 0=q 1=k 2=v
    const int b = bh / HH, h = bh % HH;

    const __bf16* Wt   = (z == 0) ? Wtq : (z == 1) ? Wtk : Wtv;
    const float*  bias = (z == 0) ? bq  : (z == 1) ? bk  : bv;

    const __bf16* Abase = xb + ((size_t)b * TT + (size_t)mt * 16) * EE;
    const __bf16* Bbase = Wt + ((size_t)h * DHH + (size_t)wv * 16) * EE;

    v8f acc = {};
#pragma unroll 4
    for (int k0 = 0; k0 < EE; k0 += 32) {
        v16bf a  = load_A(Abase + k0, EE, lane);
        v16bf bm = load_B(Bbase + k0, EE, lane);
        acc = wmma_bf16(a, bm, acc);
    }

    const int col   = lane & 15;
    const int rbase = (lane >> 4) << 3;
    const int d     = wv * 16 + col;
    const float bsv = bias[h * DHH + d];
#pragma unroll
    for (int r = 0; r < 8; ++r) {
        int t   = mt * 16 + rbase + r;
        float v = acc[r] + bsv;
        if (z == 2) {
            Vt[((size_t)bh * DHH + d) * TT + t] = (__bf16)v;
        } else {
            __bf16* O = (z == 0) ? Q : K;
            O[((size_t)bh * TT + t) * DHH + d] = (__bf16)v;
        }
    }
}

// ---------------------------------------------------------------------------
// Stage 2: causal flash attention.  One wave owns a 16-query tile.
// grid = (T/16/AW, B*H), block = 32*AW.  K/V tiles are DMA'd into per-wave
// LDS slabs by the Tensor Data Mover, double-buffered on TENSORcnt so the
// next 32-key block streams in while WMMAs + softmax run on the current one.
// ---------------------------------------------------------------------------

__global__ __launch_bounds__(32 * AW) void attn_kernel(
    const __bf16* __restrict__ Q, const __bf16* __restrict__ K,
    const __bf16* __restrict__ Vt, __bf16* __restrict__ concat) {
    __shared__ __bf16 kbuf[AW][2][32 * DHH];   // [key][dh]   per wave, 2 buffers
    __shared__ __bf16 vbuf[AW][2][DHH * 32];   // [dh][key]   per wave, 2 buffers
    __shared__ __bf16 pbuf[AW][16 * 32];       // P tile staging (C- -> A-layout)

    const int lane = threadIdx.x & 31;
    const int wv   = threadIdx.x >> 5;
    const int qt   = blockIdx.x * AW + wv;     // 0..127
    const int bh   = blockIdx.y;
    const int b = bh / HH, h = bh % HH;
    const int qbase = qt * 16;

    const __bf16* Qb = Q + ((size_t)bh * TT + qbase) * DHH;
    const v16bf aq0 = load_A(Qb,      DHH, lane);   // dh 0..31
    const v16bf aq1 = load_A(Qb + 32, DHH, lane);   // dh 32..63

    const float scale = 0.125f;   // 1/sqrt(64)
    const int col     = lane & 15;
    const int rowbase = (lane >> 4) << 3;

    float mrow[8], lrow[8];
    v8f o[4] = {};
#pragma unroll
    for (int r = 0; r < 8; ++r) { mrow[r] = -__builtin_inff(); lrow[r] = 0.0f; }

    const __bf16* Kbh  = K  + (size_t)bh * TT * DHH;
    const __bf16* Vtbh = Vt + (size_t)bh * DHH * TT;

    const int nkb = (qbase + 16 + 31) >> 5;   // 32-key blocks in causal region

    // prologue: DMA first K/V block into buffer 0
    {
        unsigned klds = (unsigned)(uintptr_t)&kbuf[wv][0][0];
        unsigned vlds = (unsigned)(uintptr_t)&vbuf[wv][0][0];
        tdm_load_2d(klds, (unsigned long long)(uintptr_t)Kbh,
                    DHH, 32, DHH, TT, DHH);
        tdm_load_2d(vlds, (unsigned long long)(uintptr_t)Vtbh,
                    32, DHH, TT, DHH, TT);
    }

    for (int kb = 0; kb < nkb; ++kb) {
        const int kb32 = kb * 32;
        const int buf  = kb & 1;
        if (kb + 1 < nkb) {
            // prefetch next block into the other buffer, then wait for current
            unsigned klds = (unsigned)(uintptr_t)&kbuf[wv][1 - buf][0];
            unsigned vlds = (unsigned)(uintptr_t)&vbuf[wv][1 - buf][0];
            tdm_load_2d(klds,
                        (unsigned long long)(uintptr_t)(Kbh + (size_t)(kb32 + 32) * DHH),
                        DHH, 32, DHH, TT, DHH);
            tdm_load_2d(vlds,
                        (unsigned long long)(uintptr_t)(Vtbh + (kb32 + 32)),
                        32, DHH, TT, DHH, TT);
            __builtin_amdgcn_s_wait_tensorcnt(2);
        } else {
            __builtin_amdgcn_s_wait_tensorcnt(0);
        }

        const __bf16* kt = &kbuf[wv][buf][0];   // [32 keys][64 dh], ld=64
        const __bf16* vt = &vbuf[wv][buf][0];   // [64 dh][32 keys], ld=32

        // S for both 16-key subtiles of this 32-key block (4 WMMAs)
        v8f s0 = {}, s1 = {};
        s0 = wmma_bf16(aq0, load_B(kt,           DHH, lane), s0);
        s0 = wmma_bf16(aq1, load_B(kt + 32,      DHH, lane), s0);
        s1 = wmma_bf16(aq0, load_B(kt + 16 * DHH,      DHH, lane), s1);
        s1 = wmma_bf16(aq1, load_B(kt + 16 * DHH + 32, DHH, lane), s1);

        // merged online-softmax stats over the full 32-key block
#pragma unroll
        for (int r = 0; r < 8; ++r) {
            const int qrow = qbase + rowbase + r;
            float a0 = s0[r] * scale;
            float a1 = s1[r] * scale;
            if (kb32 + col > qrow)      a0 = -__builtin_inff();
            if (kb32 + 16 + col > qrow) a1 = -__builtin_inff();
            float tm = fmaxf(a0, a1);
            tm = fmaxf(tm, __shfl_xor(tm, 1, 32));
            tm = fmaxf(tm, __shfl_xor(tm, 2, 32));
            tm = fmaxf(tm, __shfl_xor(tm, 4, 32));
            tm = fmaxf(tm, __shfl_xor(tm, 8, 32));
            const float mnew  = fmaxf(mrow[r], tm);
            const float alpha = __expf(mrow[r] - mnew);
            const float p0    = __expf(a0 - mnew);
            const float p1    = __expf(a1 - mnew);
            float ts = p0 + p1;
            ts += __shfl_xor(ts, 1, 32);
            ts += __shfl_xor(ts, 2, 32);
            ts += __shfl_xor(ts, 4, 32);
            ts += __shfl_xor(ts, 8, 32);
            lrow[r] = lrow[r] * alpha + ts;
            mrow[r] = mnew;
#pragma unroll
            for (int nt = 0; nt < 4; ++nt) o[nt][r] *= alpha;
            pbuf[wv][(rowbase + r) * 32 + col]      = (__bf16)p0;
            pbuf[wv][(rowbase + r) * 32 + 16 + col] = (__bf16)p1;
        }

        // P (16x32) x V (32x64): 4 WMMAs
        v16bf ap = load_A(&pbuf[wv][0], 32, lane);
#pragma unroll
        for (int nt = 0; nt < 4; ++nt) {
            o[nt] = wmma_bf16(ap, load_B(vt + nt * 16 * 32, 32, lane), o[nt]);
        }
    }

    // normalize by softmax denom and emit concat[b, t, h*64 + d] as bf16
#pragma unroll
    for (int r = 0; r < 8; ++r) {
        const float inv = 1.0f / lrow[r];
        const int t = qbase + rowbase + r;
#pragma unroll
        for (int nt = 0; nt < 4; ++nt) {
            float v = o[nt][r] * inv;
            concat[((size_t)b * TT + t) * EE + h * DHH + nt * 16 + col] = (__bf16)v;
        }
    }
}

// ---------------------------------------------------------------------------
// Stage 3: output projection GEMM  concat[4096,768] x Wo[768,768] + bo -> f32
// grid = (B*T/16, E/64), block = 128
// ---------------------------------------------------------------------------

__global__ __launch_bounds__(128) void out_proj_kernel(
    const __bf16* __restrict__ concat, const __bf16* __restrict__ Wot,
    const float* __restrict__ bo, float* __restrict__ proj) {
    const int lane = threadIdx.x & 31;
    const int wv   = threadIdx.x >> 5;
    const int mt   = blockIdx.x;
    const int nb   = blockIdx.y * 64 + wv * 16;

    const __bf16* Abase = concat + (size_t)mt * 16 * EE;
    const __bf16* Bbase = Wot + (size_t)nb * EE;

    v8f acc = {};
#pragma unroll 4
    for (int k0 = 0; k0 < EE; k0 += 32) {
        v16bf a  = load_A(Abase + k0, EE, lane);
        v16bf bm = load_B(Bbase + k0, EE, lane);
        acc = wmma_bf16(a, bm, acc);
    }

    const int col = lane & 15, rbase = (lane >> 4) << 3;
    const float bias = bo[nb + col];
#pragma unroll
    for (int r = 0; r < 8; ++r) {
        size_t row = (size_t)mt * 16 + rbase + r;
        proj[row * EE + nb + col] = acc[r] + bias;
    }
}

// ---------------------------------------------------------------------------
// Stage 4: LayerNorm(proj) + residual  (f32, bandwidth trivial)
// ---------------------------------------------------------------------------

__global__ __launch_bounds__(256) void ln_res_kernel(
    const float* __restrict__ proj, const float* __restrict__ x,
    const float* __restrict__ gamma, const float* __restrict__ beta,
    float* __restrict__ out) {
    const int row = blockIdx.x;
    const float* p  = proj + (size_t)row * EE;
    const float* xr = x + (size_t)row * EE;
    float* orow = out + (size_t)row * EE;

    float s = 0.0f, ss = 0.0f;
    for (int i = threadIdx.x; i < EE; i += 256) {
        float v = p[i];
        s += v;
        ss += v * v;
    }
    __shared__ float sh[16];
#pragma unroll
    for (int m = 1; m < 32; m <<= 1) {
        s  += __shfl_xor(s, m, 32);
        ss += __shfl_xor(ss, m, 32);
    }
    const int lane = threadIdx.x & 31, wv = threadIdx.x >> 5;
    if (lane == 0) { sh[wv] = s; sh[8 + wv] = ss; }
    __syncthreads();
    if (threadIdx.x < 32) {
        float a  = (threadIdx.x < 8) ? sh[threadIdx.x] : 0.0f;
        float aa = (threadIdx.x < 8) ? sh[8 + threadIdx.x] : 0.0f;
#pragma unroll
        for (int m = 1; m < 8; m <<= 1) {
            a  += __shfl_xor(a, m, 32);
            aa += __shfl_xor(aa, m, 32);
        }
        if (threadIdx.x == 0) { sh[0] = a; sh[1] = aa; }
    }
    __syncthreads();
    const float mu   = sh[0] * (1.0f / EE);
    const float var  = sh[1] * (1.0f / EE) - mu * mu;
    const float rstd = rsqrtf(var + 1e-5f);
    for (int i = threadIdx.x; i < EE; i += 256) {
        float v = (p[i] - mu) * rstd * gamma[i] + beta[i];
        orow[i] = xr[i] + v;
    }
}

// ---------------------------------------------------------------------------
// Host launcher
// ---------------------------------------------------------------------------

extern "C" void kernel_launch(void* const* d_in, const int* in_sizes, int n_in,
                              void* d_out, int out_size, void* d_ws, size_t ws_size,
                              hipStream_t stream) {
    (void)in_sizes; (void)n_in; (void)out_size; (void)ws_size;
    const float* x     = (const float*)d_in[0];
    const float* Wq    = (const float*)d_in[1];
    const float* bq    = (const float*)d_in[2];
    const float* Wk    = (const float*)d_in[3];
    const float* bk    = (const float*)d_in[4];
    const float* Wv    = (const float*)d_in[5];
    const float* bv    = (const float*)d_in[6];
    const float* Wo    = (const float*)d_in[7];
    const float* bo    = (const float*)d_in[8];
    const float* gamma = (const float*)d_in[9];
    const float* beta  = (const float*)d_in[10];
    float* out = (float*)d_out;

    // workspace carve-up (~49 MB total)
    char* w = (char*)d_ws;
    __bf16* xb  = (__bf16*)w; w += (size_t)2 * BB * TT * EE;
    __bf16* Wtq = (__bf16*)w; w += (size_t)2 * HH * DHH * EE;
    __bf16* Wtk = (__bf16*)w; w += (size_t)2 * HH * DHH * EE;
    __bf16* Wtv = (__bf16*)w; w += (size_t)2 * HH * DHH * EE;
    __bf16* Wot = (__bf16*)w; w += (size_t)2 * EE * EE;
    __bf16* Qb  = (__bf16*)w; w += (size_t)2 * BB * HH * TT * DHH;
    __bf16* Kb  = (__bf16*)w; w += (size_t)2 * BB * HH * TT * DHH;
    __bf16* Vt  = (__bf16*)w; w += (size_t)2 * BB * HH * TT * DHH;
    __bf16* cc  = (__bf16*)w; w += (size_t)2 * BB * TT * EE;
    float*  pj  = (float*)w;

    const int nx = BB * TT * EE;
    cvt_x_kernel<<<(nx + 255) / 256, 256, 0, stream>>>(x, xb, nx);
    const int nw = HH * EE * DHH;
    cvt_w_kernel<<<(nw + 255) / 256, 256, 0, stream>>>(Wq, Wtq);
    cvt_w_kernel<<<(nw + 255) / 256, 256, 0, stream>>>(Wk, Wtk);
    cvt_w_kernel<<<(nw + 255) / 256, 256, 0, stream>>>(Wv, Wtv);
    const int nwo = EE * EE;
    cvt_wo_kernel<<<(nwo + 255) / 256, 256, 0, stream>>>(Wo, Wot);

    qkv_proj_kernel<<<dim3(TT / 16, BB * HH, 3), 128, 0, stream>>>(
        xb, Wtq, Wtk, Wtv, bq, bk, bv, Qb, Kb, Vt);

    attn_kernel<<<dim3(TT / 16 / AW, BB * HH), 32 * AW, 0, stream>>>(Qb, Kb, Vt, cc);

    out_proj_kernel<<<dim3(BB * TT / 16, EE / 64), 128, 0, stream>>>(cc, Wot, bo, pj);

    ln_res_kernel<<<BB * TT, 256, 0, stream>>>(pj, x, gamma, beta, out);
}
